// RegionProposalNetwork_80367428043457
// MI455X (gfx1250) — matrix operand order repaired and checked by reading, not compiled, not run on hardware
//
#include <hip/hip_runtime.h>
#include <hip/hip_bf16.h>
#include <math.h>

// ---------------------------------------------------------------------------
// RPN forward for MI455X (gfx1250, wave32, WMMA).
// Conv 3x3 768->768 on 14x14 expressed as GEMM M=208(pad of 196), N=768,
// K=6912 using v_wmma_f32_16x16x32_f16 (f32 accumulate). Everything else is
// small VALU/LDS work (heads, decode, O(n^2) top-k rank selection, NMS).
// ---------------------------------------------------------------------------

#define IN_CHN   768
#define FHW      14
#define NPOS     196          // 14*14
#define ANUM     9
#define NANCH    1764         // 196*9
#define KDIM     6912         // 768*9
#define MPAD     208          // 13*16
#define PRENMS   1000
#define TOPK_OUT 300
#define NMS_TH   0.7f
#define MIN_SZ   16.0f
#define IMG_SZ   224.0f
#define BBOX_CLIP 4.135166556742356f   // log(1000/16)

typedef __attribute__((ext_vector_type(16))) _Float16 v16h;
typedef __attribute__((ext_vector_type(8)))  _Float16 v8h;
typedef __attribute__((ext_vector_type(8)))  float    v8f;

// ---- 1. w_conv (O,I,3,3) f32 -> f16, already B^T im2col layout [o][k] -----
__global__ void rpn_pack_w_kernel(const float* __restrict__ w_conv,
                                  _Float16* __restrict__ Bt) {
  int idx = blockIdx.x * 256 + threadIdx.x;
  if (idx < IN_CHN * KDIM) Bt[idx] = (_Float16)w_conv[idx];
}

// ---- 2. im2col of feature (drop CLS token) -> f16 A[MPAD][KDIM] ----------
// feat[c][h][w] == feature_flat[768 + c*196 + h*14 + w]
__global__ void rpn_im2col_kernel(const float* __restrict__ feature,
                                  _Float16* __restrict__ Af) {
  int idx = blockIdx.x * 256 + threadIdx.x;
  if (idx >= MPAD * KDIM) return;
  int m = idx / KDIM, k = idx % KDIM;
  float v = 0.0f;
  if (m < NPOS) {
    int ph = m / FHW, pw = m % FHW;
    int ci = k / 9, off = k % 9;
    int y = ph + off / 3 - 1;
    int x = pw + off % 3 - 1;
    if (y >= 0 && y < FHW && x >= 0 && x < FHW)
      v = feature[IN_CHN + ci * NPOS + y * FHW + x];
  }
  Af[idx] = (_Float16)v;
}

// ---- 3. Conv-as-GEMM via WMMA; bias + ReLU fused on store ----------------
// One wave per 16x16 C tile; 4 waves per block. Fragment layouts use contiguous K
// (not the ISA table order; A and B match): A 16x32 f16 per-lane K = 16g + e (e<16), B 32x16
// f16 with K=0..15 in lanes 0-15, K=16..31 in lanes 16-31 (Bt row-gather).
__global__ void __launch_bounds__(128)
rpn_conv_wmma_kernel(const _Float16* __restrict__ Af,
                     const _Float16* __restrict__ Bt,
                     const float* __restrict__ b_conv,
                     float* __restrict__ xbuf) {
  const int lane = threadIdx.x & 31;
  const int wave = threadIdx.x >> 5;
  const int m0 = blockIdx.x * 16;
  const int n0 = (blockIdx.y * 4 + wave) * 16;
  const int g = lane >> 4;     // half-wave group
  const int r = lane & 15;

  const _Float16* __restrict__ Arow = Af + (size_t)(m0 + r) * KDIM;
  const _Float16* __restrict__ Brow = Bt + (size_t)(n0 + r) * KDIM;

  v8f c = {};
  for (int kk = 0; kk < KDIM; kk += 32) {
    v8h alo = *(const v8h*)(Arow + kk + 16 * g);
    v8h ahi = *(const v8h*)(Arow + kk + 16 * g + 8);
    v16h a, b;
#pragma unroll
    for (int e = 0; e < 8; ++e) { a[e] = alo[e]; a[8 + e] = ahi[e]; }
    b = *(const v16h*)(Brow + kk + 16 * g);
    // D(f32 16x16) = A(16x32 f16) * B(32x16 f16) + C
    c = __builtin_amdgcn_wmma_f32_16x16x32_f16(
        /*neg_a=*/false, a, /*neg_b=*/false, b,
        /*c_mod=*/(short)0, c, /*reuse_a=*/false, /*reuse_b=*/false);
  }

  const int col = n0 + r;
  const float bias = b_conv[col];
#pragma unroll
  for (int j = 0; j < 8; ++j) {
    int row = m0 + 8 * g + j;          // C layout: VGPR j -> M = j + 8g
    if (row < NPOS) {
      float v = c[j] + bias;
      xbuf[row * IN_CHN + col] = v > 0.0f ? v : 0.0f;
    }
  }
}

// ---- 4. 1x1 heads: cls (9) and reg (36) ----------------------------------
__global__ void rpn_heads_kernel(const float* __restrict__ xbuf,
                                 const float* __restrict__ w_cls,
                                 const float* __restrict__ b_cls,
                                 const float* __restrict__ w_reg,
                                 const float* __restrict__ b_reg,
                                 float* __restrict__ scores,
                                 float* __restrict__ regout) {
  int t = blockIdx.x * 128 + threadIdx.x;
  if (t >= NPOS * 45) return;
  int p = t / 45, o = t % 45;
  const float* __restrict__ xr = xbuf + p * IN_CHN;
  const float* __restrict__ wr =
      (o < ANUM) ? (w_cls + o * IN_CHN) : (w_reg + (o - ANUM) * IN_CHN);
  float acc = 0.0f;
  for (int c0 = 0; c0 < IN_CHN; ++c0) acc += xr[c0] * wr[c0];
  if (o < ANUM) scores[p * ANUM + o] = acc + b_cls[o];
  else          regout[p * 36 + (o - ANUM)] = acc + b_reg[o - ANUM];
}

// ---- 5. anchors + box decode + sigmoid -----------------------------------
__global__ void rpn_decode_kernel(const float* __restrict__ scores,
                                  const float* __restrict__ regout,
                                  float* __restrict__ proposals,
                                  float* __restrict__ probs) {
  int q = blockIdx.x * 256 + threadIdx.x;
  if (q >= NANCH) return;
  int p = q / ANUM, a = q % ANUM;
  int hy = p / FHW, wx = p % FHW;
  const float ratios[3] = {0.5f, 1.0f, 2.0f};
  const float scales[3] = {128.0f, 256.0f, 512.0f};
  float hr = sqrtf(ratios[a / 3]);
  float wr = 1.0f / hr;
  float wsz = wr * scales[a % 3];
  float hsz = hr * scales[a % 3];
  float bx0 = rintf(-wsz * 0.5f), by0 = rintf(-hsz * 0.5f);
  float bx2 = rintf( wsz * 0.5f), by2 = rintf( hsz * 0.5f);
  float ax0 = wx * 16.0f + bx0, ay0 = hy * 16.0f + by0;
  float ax2 = wx * 16.0f + bx2, ay2 = hy * 16.0f + by2;
  float w = ax2 - ax0, h = ay2 - ay0;
  float cx = ax0 + 0.5f * w, cy = ay0 + 0.5f * h;
  const float* d = regout + p * 36 + a * 4;
  float dx = d[0], dy = d[1];
  float dw = fminf(d[2], BBOX_CLIP), dh = fminf(d[3], BBOX_CLIP);
  float pcx = dx * w + cx, pcy = dy * h + cy;
  float pw = expf(dw) * w, ph = expf(dh) * h;
  proposals[q * 4 + 0] = pcx - 0.5f * pw;
  proposals[q * 4 + 1] = pcy - 0.5f * ph;
  proposals[q * 4 + 2] = pcx + 0.5f * pw;
  proposals[q * 4 + 3] = pcy + 0.5f * ph;
  probs[q] = 1.0f / (1.0f + expf(-scores[q]));
}

// ---- 6. rank-based top-1000, clip, min-size ------------------------------
// rank(i) = #{j: p[j] > p[i]} + #{j < i: p[j] == p[i]}  (jax top_k semantics)
__global__ void rpn_topk1_kernel(const float* __restrict__ probs,
                                 const float* __restrict__ proposals,
                                 float* __restrict__ boxesS,
                                 float* __restrict__ scoresS,
                                 int* __restrict__ validS) {
  __shared__ float sp[NANCH];
  for (int i = threadIdx.x; i < NANCH; i += blockDim.x) sp[i] = probs[i];
  __syncthreads();
  int i = blockIdx.x * blockDim.x + threadIdx.x;
  if (i >= NANCH) return;
  float v = sp[i];
  int rank = 0;
  for (int j = 0; j < NANCH; ++j) {
    float u = sp[j];
    rank += (u > v) || (u == v && j < i);
  }
  if (rank < PRENMS) {
    float x0 = fminf(fmaxf(proposals[i * 4 + 0], 0.0f), IMG_SZ);
    float y0 = fminf(fmaxf(proposals[i * 4 + 1], 0.0f), IMG_SZ);
    float x1 = fminf(fmaxf(proposals[i * 4 + 2], 0.0f), IMG_SZ);
    float y1 = fminf(fmaxf(proposals[i * 4 + 3], 0.0f), IMG_SZ);
    boxesS[rank * 4 + 0] = x0;
    boxesS[rank * 4 + 1] = y0;
    boxesS[rank * 4 + 2] = x1;
    boxesS[rank * 4 + 3] = y1;
    scoresS[rank] = v;
    validS[rank] = (x1 - x0 >= MIN_SZ) && (y1 - y0 >= MIN_SZ);
  }
}

// ---- 7. greedy NMS, single block, boxes + keep mask in LDS ---------------
__global__ void __launch_bounds__(1024)
rpn_nms_kernel(const float* __restrict__ boxesS,
               const int* __restrict__ validS,
               int* __restrict__ keepS) {
  __shared__ float bx[PRENMS][4];
  __shared__ int keep[PRENMS];
  int t = threadIdx.x;
  for (int i = t; i < PRENMS; i += 1024) {
    bx[i][0] = boxesS[i * 4 + 0];
    bx[i][1] = boxesS[i * 4 + 1];
    bx[i][2] = boxesS[i * 4 + 2];
    bx[i][3] = boxesS[i * 4 + 3];
    keep[i] = validS[i];
  }
  __syncthreads();
  float a0 = 0, a1 = 0, a2 = 0, a3 = 0, area_t = 0;
  if (t < PRENMS) {
    a0 = bx[t][0]; a1 = bx[t][1]; a2 = bx[t][2]; a3 = bx[t][3];
    area_t = (a2 - a0) * (a3 - a1);
  }
  for (int i = 0; i < PRENMS - 1; ++i) {
    if (keep[i] && t > i && t < PRENMS && keep[t]) {
      float ix0 = fmaxf(a0, bx[i][0]);
      float iy0 = fmaxf(a1, bx[i][1]);
      float ix1 = fminf(a2, bx[i][2]);
      float iy1 = fminf(a3, bx[i][3]);
      float iw = fmaxf(ix1 - ix0, 0.0f);
      float ih = fmaxf(iy1 - iy0, 0.0f);
      float inter = iw * ih;
      float areai = (bx[i][2] - bx[i][0]) * (bx[i][3] - bx[i][1]);
      float uni = area_t + areai - inter;
      float iou = uni > 0.0f ? inter / uni : 0.0f;
      if (iou > NMS_TH) keep[t] = 0;
    }
    __syncthreads();
  }
  for (int i = t; i < PRENMS; i += 1024) keepS[i] = keep[i];
}

// ---- 8. rank-based top-300 + pad masking into d_out ----------------------
__global__ void rpn_topk2_kernel(const float* __restrict__ boxesS,
                                 const float* __restrict__ scoresS,
                                 const int* __restrict__ keepS,
                                 float* __restrict__ out) {
  __shared__ float sm[PRENMS];
  for (int i = threadIdx.x; i < PRENMS; i += blockDim.x)
    sm[i] = keepS[i] ? scoresS[i] : -1.0f;
  __syncthreads();
  int i = blockIdx.x * blockDim.x + threadIdx.x;
  if (i >= PRENMS) return;
  float v = sm[i];
  int rank = 0;
  for (int j = 0; j < PRENMS; ++j) {
    float u = sm[j];
    rank += (u > v) || (u == v && j < i);
  }
  if (rank < TOPK_OUT) {
    float pad = v > 0.0f ? 1.0f : 0.0f;
    out[TOPK_OUT * 4 + rank] = v * pad;               // scores at [1200,1500)
    out[rank * 4 + 0] = boxesS[i * 4 + 0] * pad;      // boxes at [0,1200)
    out[rank * 4 + 1] = boxesS[i * 4 + 1] * pad;
    out[rank * 4 + 2] = boxesS[i * 4 + 2] * pad;
    out[rank * 4 + 3] = boxesS[i * 4 + 3] * pad;
  }
}

extern "C" void kernel_launch(void* const* d_in, const int* in_sizes, int n_in,
                              void* d_out, int out_size, void* d_ws, size_t ws_size,
                              hipStream_t stream) {
  (void)in_sizes; (void)n_in; (void)out_size; (void)ws_size;
  // inputs: 0 image, 1 feature, 2 w_conv, 3 b_conv, 4 w_cls, 5 b_cls, 6 w_reg, 7 b_reg
  const float* feature = (const float*)d_in[1];
  const float* w_conv  = (const float*)d_in[2];
  const float* b_conv  = (const float*)d_in[3];
  const float* w_cls   = (const float*)d_in[4];
  const float* b_cls   = (const float*)d_in[5];
  const float* w_reg   = (const float*)d_in[6];
  const float* b_reg   = (const float*)d_in[7];
  float* out = (float*)d_out;

  // workspace carve-up (256B-aligned)
  char* ws = (char*)d_ws;
  size_t off = 0;
  auto align256 = [](size_t x) { return (x + 255) & ~(size_t)255; };
  _Float16* Af = (_Float16*)(ws + off); off += align256((size_t)MPAD * KDIM * 2);
  _Float16* Bt = (_Float16*)(ws + off); off += align256((size_t)IN_CHN * KDIM * 2);
  float* xbuf      = (float*)(ws + off); off += align256((size_t)NPOS * IN_CHN * 4);
  float* scores    = (float*)(ws + off); off += align256((size_t)NANCH * 4);
  float* regout    = (float*)(ws + off); off += align256((size_t)NPOS * 36 * 4);
  float* proposals = (float*)(ws + off); off += align256((size_t)NANCH * 4 * 4);
  float* probs     = (float*)(ws + off); off += align256((size_t)NANCH * 4);
  float* boxesS    = (float*)(ws + off); off += align256((size_t)PRENMS * 4 * 4);
  float* scoresS   = (float*)(ws + off); off += align256((size_t)PRENMS * 4);
  int*   keepS     = (int*)(ws + off);   off += align256((size_t)PRENMS * 4);
  int*   validS    = (int*)(ws + off);   off += align256((size_t)PRENMS * 4);

  // 1. pack weights to f16 (B^T im2col layout)
  rpn_pack_w_kernel<<<(IN_CHN * KDIM) / 256, 256, 0, stream>>>(w_conv, Bt);
  // 2. im2col feature -> f16 A
  rpn_im2col_kernel<<<(MPAD * KDIM) / 256, 256, 0, stream>>>(feature, Af);
  // 3. WMMA GEMM: grid (13 M-tiles, 12 N-groups of 4 waves)
  rpn_conv_wmma_kernel<<<dim3(MPAD / 16, IN_CHN / 64), 128, 0, stream>>>(
      Af, Bt, b_conv, xbuf);
  // 4. heads
  rpn_heads_kernel<<<(NPOS * 45 + 127) / 128, 128, 0, stream>>>(
      xbuf, w_cls, b_cls, w_reg, b_reg, scores, regout);
  // 5. decode + sigmoid
  rpn_decode_kernel<<<(NANCH + 255) / 256, 256, 0, stream>>>(
      scores, regout, proposals, probs);
  // 6. top-1000 (rank select), clip, min-size
  rpn_topk1_kernel<<<(NANCH + 255) / 256, 256, 0, stream>>>(
      probs, proposals, boxesS, scoresS, validS);
  // 7. NMS (single block, sequential pivots)
  rpn_nms_kernel<<<1, 1024, 0, stream>>>(boxesS, validS, keepS);
  // 8. top-300 + pad masking into d_out
  rpn_topk2_kernel<<<(PRENMS + 255) / 256, 256, 0, stream>>>(
      boxesS, scoresS, keepS, out);
}
